// CodebookLearner_13846974562840
// MI455X (gfx1250) — compile-verified, matrix operand order checked by reading, not run on hardware
//
#include <hip/hip_runtime.h>
#include <hip/hip_bf16.h>
#include <math.h>

// ---------------------------------------------------------------------------
// Gumbel-softmax VQ codebook learner, fused for MI455X (gfx1250, wave32).
//   B=4096, D=512, K=4096, T=8 steps.
//   grid = 256 WGs (16 B-rows each), 256 threads (8 waves) per WG.
//   GEMMs via v_wmma_f32_16x16x32_bf16; logits/y staged in 313KB of LDS.
//   Streaming softmax passes vectorized to b128 traffic (memory-bound side).
// ---------------------------------------------------------------------------

typedef __attribute__((ext_vector_type(16))) __bf16 v16bf;
typedef __attribute__((ext_vector_type(8)))  __bf16 v8bf;
typedef __attribute__((ext_vector_type(8)))  float  v8f;
typedef __attribute__((ext_vector_type(4)))  float  v4f;

#define B_    4096
#define D_    512
#define K_    4096
#define T_    8
#define EPSF  1e-10f

#define MTILE 16
#define LROW  4100   // logits row stride (floats); 4100%64=4 -> rows hit distinct banks
#define CROW  516    // contrib row stride (floats); rows 16B aligned
#define RBROW 520    // residual-bf16 row stride (bf16); 1040B rows, 16B aligned

static __device__ __forceinline__ __bf16 f2bf(float f) {
  return (__bf16)f;   // native RNE convert on gfx1250 (v_cvt_*_bf16_f32)
}

static __device__ __forceinline__ v16bf cat8(v8bf lo, v8bf hi) {
  return __builtin_shufflevector(lo, hi, 0,1,2,3,4,5,6,7,8,9,10,11,12,13,14,15);
}

static __device__ __forceinline__ v8f wmma_bf16(v16bf a, v16bf b, v8f c) {
  return __builtin_amdgcn_wmma_f32_16x16x32_bf16(false, a, false, b, (short)0, c,
                                                 false, false);
}

// ---------------------------------------------------------------------------
// prep: codebook f32[K][D] -> bf16 [K][D] and bf16 transposed [D][K] in d_ws.
// Also zeroes the two global loss accumulators (stream-ordered before main).
// ---------------------------------------------------------------------------
__global__ void prep_kernel(const float* __restrict__ cb,
                            __bf16* __restrict__ cbb,
                            __bf16* __restrict__ cbt,
                            float* __restrict__ acc) {
  int idx = blockIdx.x * 256 + threadIdx.x;          // 0 .. K*D-1
  if (blockIdx.x == 0 && threadIdx.x < 2) acc[threadIdx.x] = 0.0f;
  int k = idx >> 9;        // / D_
  int d = idx & (D_ - 1);
  __bf16 b = f2bf(cb[idx]);
  cbb[idx] = b;
  cbt[(size_t)d * K_ + k] = b;
}

// ---------------------------------------------------------------------------
// main fused kernel
// ---------------------------------------------------------------------------
__global__ void __launch_bounds__(256, 1)
gumbel_vq_kernel(const float*  __restrict__ targets,   // [B][D]
                 const float*  __restrict__ noise,     // [T][B][K]
                 const __bf16* __restrict__ cbb,       // [K][D] bf16
                 const __bf16* __restrict__ cbt,       // [D][K] bf16
                 float* __restrict__ out_recon,        // [B][D]
                 float* __restrict__ out_dists,        // [T][B][K]
                 float* __restrict__ acc)              // [2]: {mse_sum, ent_sum}
{
  extern __shared__ char smem_raw[];
  float*  logits  = (float*)smem_raw;                  // 16 x LROW  f32
  float*  contrib = logits + MTILE * LROW;             // 16 x CROW  f32
  __bf16* resbf   = (__bf16*)(contrib + MTILE * CROW); // 16 x RBROW bf16
  float*  scr     = (float*)(resbf + MTILE * RBROW);   // 256 f32
  float*  rowstat = scr + 256;                         // 32  f32

  const int tid  = threadIdx.x;
  const int ln   = tid & 15;          // fragment lane-in-half
  const int hi   = (tid >> 4) & 1;    // upper half of wave
  const int wave = tid >> 5;          // 0..7
  const int b0   = blockIdx.x * MTILE;

  // each thread persistently owns row m, 32 D-columns starting at db
  const int m  = tid >> 4;            // 0..15
  const int db = (tid & 15) * 32;     // 0..480
  float res[32], rec[32];

  { // init residual = targets, recon = 0
    const float* trow = targets + (size_t)(b0 + m) * D_ + db;
    #pragma unroll
    for (int i = 0; i < 8; ++i) {
      v4f v = *(const v4f*)(trow + 4 * i);
      res[4*i+0] = v[0]; res[4*i+1] = v[1]; res[4*i+2] = v[2]; res[4*i+3] = v[3];
    }
    #pragma unroll
    for (int i = 0; i < 32; ++i) rec[i] = 0.0f;
    #pragma unroll
    for (int g = 0; g < 4; ++g) {
      v8bf p;
      #pragma unroll
      for (int i = 0; i < 8; ++i) p[i] = f2bf(res[g*8+i]);
      *(v8bf*)(resbf + m * RBROW + db + g * 8) = p;
    }
  }
  __syncthreads();

  float ent_acc = 0.0f;
  float decay   = 1.0f;

  for (int t = 0; t < T_; ++t) {
    // ---- GEMM1: logits[16][4096] = residual[16][512] @ codebook^T ----------
    {
      const int wb = wave * 512;                      // this wave's K-columns
      for (int ng = 0; ng < 8; ++ng) {                // 4 tiles of 16 cols each
        v8f a0 = {}, a1 = {}, a2 = {}, a3 = {};
        #pragma unroll 4
        for (int kk = 0; kk < 16; ++kk) {             // D in chunks of 32
          const int d0 = kk * 32 + hi * 8;
          const __bf16* ap = resbf + ln * RBROW + d0;
          v16bf afrag = cat8(*(const v8bf*)ap, *(const v8bf*)(ap + 16));
          const int nb = wb + ng * 64;
          const __bf16* bp0 = cbb + (size_t)(nb +  0 + ln) * D_ + d0;
          const __bf16* bp1 = cbb + (size_t)(nb + 16 + ln) * D_ + d0;
          const __bf16* bp2 = cbb + (size_t)(nb + 32 + ln) * D_ + d0;
          const __bf16* bp3 = cbb + (size_t)(nb + 48 + ln) * D_ + d0;
          a0 = wmma_bf16(afrag, cat8(*(const v8bf*)bp0, *(const v8bf*)(bp0 + 16)), a0);
          a1 = wmma_bf16(afrag, cat8(*(const v8bf*)bp1, *(const v8bf*)(bp1 + 16)), a1);
          a2 = wmma_bf16(afrag, cat8(*(const v8bf*)bp2, *(const v8bf*)(bp2 + 16)), a2);
          a3 = wmma_bf16(afrag, cat8(*(const v8bf*)bp3, *(const v8bf*)(bp3 + 16)), a3);
        }
        v8f av[4] = {a0, a1, a2, a3};
        #pragma unroll
        for (int j = 0; j < 4; ++j) {
          const int col = wb + ng * 64 + j * 16 + ln;
          #pragma unroll
          for (int e = 0; e < 8; ++e)
            logits[(e + hi * 8) * LROW + col] = av[j][e];
        }
      }
    }
    __syncthreads();

    // ---- softmax over K with Gumbel noise ----------------------------------
    // row r handled by 16 threads; each owns 64 aligned float4 chunks
    const int r = m;
    const size_t growbase = ((size_t)t * B_ + b0 + r) * (size_t)K_;
    const float* nrow = noise + growbase;
    float* lrow = logits + r * LROW;

    // pass 1: add gumbel, row max  (b128 NT noise loads, b128 LDS rmw)
    float mx = -3.4e38f;
    #pragma unroll 4
    for (int i = 0; i < 64; ++i) {
      const int c = ln * 4 + i * 64;
      v4f u = __builtin_nontemporal_load((const v4f*)(nrow + c));
      v4f l = *(const v4f*)(lrow + c);
      #pragma unroll
      for (int e = 0; e < 4; ++e) {
        float g = -__logf(-__logf(u[e] + EPSF) + EPSF);
        l[e] += g;
        mx = fmaxf(mx, l[e]);
      }
      *(v4f*)(lrow + c) = l;
    }
    scr[r * 16 + ln] = mx;
    __syncthreads();
    if (tid < 16) {
      float v = scr[tid * 16];
      #pragma unroll
      for (int i = 1; i < 16; ++i) v = fmaxf(v, scr[tid * 16 + i]);
      rowstat[tid] = v;
    }
    __syncthreads();

    // pass 2: exp + row sum
    const float rmax = rowstat[r];
    float sum = 0.0f;
    #pragma unroll 4
    for (int i = 0; i < 64; ++i) {
      const int c = ln * 4 + i * 64;
      v4f l = *(const v4f*)(lrow + c);
      #pragma unroll
      for (int e = 0; e < 4; ++e) { l[e] = __expf(l[e] - rmax); sum += l[e]; }
      *(v4f*)(lrow + c) = l;
    }
    scr[r * 16 + ln] = sum;
    __syncthreads();
    if (tid < 16) {
      float v = 0.0f;
      #pragma unroll
      for (int i = 0; i < 16; ++i) v += scr[tid * 16 + i];
      rowstat[16 + tid] = v;
    }
    __syncthreads();

    // pass 3: normalize, entropy partial, stream dists out (NT scalar stores:
    // dists base is offset by the scalar loss slot so 16B stores would misalign)
    const float rinv = 1.0f / rowstat[16 + r];
    float* drow = out_dists + growbase;
    #pragma unroll 4
    for (int i = 0; i < 64; ++i) {
      const int c = ln * 4 + i * 64;
      v4f l = *(const v4f*)(lrow + c);
      #pragma unroll
      for (int e = 0; e < 4; ++e) {
        float y = l[e] * rinv;
        l[e] = y;
        ent_acc -= y * __logf(y + EPSF);
        __builtin_nontemporal_store(y, drow + c + e);
      }
      *(v4f*)(lrow + c) = l;
    }
    __syncthreads();

    // ---- GEMM2: contrib[16][512] = y[16][4096] @ codebook ------------------
    {
      const int wd = wave * 64;                       // this wave's D-columns
      v8f a0 = {}, a1 = {}, a2 = {}, a3 = {};
      for (int kk = 0; kk < 128; ++kk) {              // K in chunks of 32
        const int k0 = kk * 32 + hi * 8;
        const float* yr = logits + ln * LROW + k0;
        v4f y0 = *(const v4f*)(yr);
        v4f y1 = *(const v4f*)(yr + 4);
        v4f y2 = *(const v4f*)(yr + 16);
        v4f y3 = *(const v4f*)(yr + 20);
        v16bf afrag;
        #pragma unroll
        for (int e = 0; e < 4; ++e) {
          afrag[e +  0] = f2bf(y0[e]);
          afrag[e +  4] = f2bf(y1[e]);
          afrag[e +  8] = f2bf(y2[e]);
          afrag[e + 12] = f2bf(y3[e]);
        }
        const __bf16* bp0 = cbt + (size_t)(wd +  0 + ln) * K_ + k0;
        const __bf16* bp1 = cbt + (size_t)(wd + 16 + ln) * K_ + k0;
        const __bf16* bp2 = cbt + (size_t)(wd + 32 + ln) * K_ + k0;
        const __bf16* bp3 = cbt + (size_t)(wd + 48 + ln) * K_ + k0;
        a0 = wmma_bf16(afrag, cat8(*(const v8bf*)bp0, *(const v8bf*)(bp0 + 16)), a0);
        a1 = wmma_bf16(afrag, cat8(*(const v8bf*)bp1, *(const v8bf*)(bp1 + 16)), a1);
        a2 = wmma_bf16(afrag, cat8(*(const v8bf*)bp2, *(const v8bf*)(bp2 + 16)), a2);
        a3 = wmma_bf16(afrag, cat8(*(const v8bf*)bp3, *(const v8bf*)(bp3 + 16)), a3);
      }
      v8f av[4] = {a0, a1, a2, a3};
      #pragma unroll
      for (int j = 0; j < 4; ++j) {
        const int col = wd + j * 16 + ln;
        #pragma unroll
        for (int e = 0; e < 8; ++e)
          contrib[(e + hi * 8) * CROW + col] = decay * av[j][e];
      }
    }
    __syncthreads();

    // ---- residual / recon update (register-resident, b128 LDS traffic) -----
    #pragma unroll
    for (int g = 0; g < 8; ++g) {
      v4f c = *(const v4f*)(contrib + m * CROW + db + g * 4);
      #pragma unroll
      for (int e = 0; e < 4; ++e) {
        res[g*4+e] -= c[e];
        rec[g*4+e] += c[e];
      }
    }
    #pragma unroll
    for (int g = 0; g < 4; ++g) {
      v8bf p;
      #pragma unroll
      for (int i = 0; i < 8; ++i) p[i] = f2bf(res[g*8+i]);
      *(v8bf*)(resbf + m * RBROW + db + g * 8) = p;
    }
    decay *= 0.9f;
    __syncthreads();
  }

  // ---- write reconstruction; per-thread loss partials -----------------------
  {
    float* orow = out_recon + (size_t)(b0 + m) * D_ + db;
    #pragma unroll
    for (int i = 0; i < 8; ++i) {
      v4f v;
      v[0] = rec[4*i+0]; v[1] = rec[4*i+1]; v[2] = rec[4*i+2]; v[3] = rec[4*i+3];
      *(v4f*)(orow + 4 * i) = v;
    }
    const float* trow = targets + (size_t)(b0 + m) * D_ + db;
    float se = 0.0f;
    #pragma unroll
    for (int i = 0; i < 8; ++i) {
      v4f v = *(const v4f*)(trow + 4 * i);
      #pragma unroll
      for (int e = 0; e < 4; ++e) { float d = rec[4*i+e] - v[e]; se += d * d; }
    }
    scr[tid]     = se;        // mse partials
    contrib[tid] = ent_acc;   // reuse contrib as entropy scratch
  }
  __syncthreads();
  for (int s = 128; s > 0; s >>= 1) {
    if (tid < s) { scr[tid] += scr[tid + s]; contrib[tid] += contrib[tid + s]; }
    __syncthreads();
  }
  if (tid == 0) {
    atomicAdd(&acc[0], scr[0]);
    atomicAdd(&acc[1], contrib[0]);
  }
}

// ---------------------------------------------------------------------------
__global__ void finalize_kernel(const float* __restrict__ acc,
                                float* __restrict__ out_loss) {
  if (blockIdx.x == 0 && threadIdx.x == 0) {
    float mse = acc[0] / (float)((size_t)B_ * D_);
    float ent = acc[1] / (float)((size_t)T_ * B_);
    *out_loss = mse + 0.01f * ent;
  }
}

// ---------------------------------------------------------------------------
extern "C" void kernel_launch(void* const* d_in, const int* in_sizes, int n_in,
                              void* d_out, int out_size, void* d_ws, size_t ws_size,
                              hipStream_t stream) {
  (void)in_sizes; (void)n_in; (void)out_size; (void)ws_size;
  const float* targets  = (const float*)d_in[0];
  const float* codebook = (const float*)d_in[1];
  const float* noise    = (const float*)d_in[2];
  float* out = (float*)d_out;

  char* ws = (char*)d_ws;
  float*  acc = (float*)ws;                                  // 2 f32 (16B slot)
  __bf16* cbb = (__bf16*)(ws + 16);                          // [K][D] bf16
  __bf16* cbt = (__bf16*)(ws + 16 + (size_t)K_ * D_ * 2);    // [D][K] bf16

  prep_kernel<<<(K_ * D_) / 256, 256, 0, stream>>>(codebook, cbb, cbt, acc);

  const size_t smem = (size_t)MTILE * LROW * 4   // logits
                    + (size_t)MTILE * CROW * 4   // contrib
                    + (size_t)MTILE * RBROW * 2  // residual bf16
                    + 256 * 4 + 32 * 4;          // scr + rowstat  (= 313216 B)

  float* out_recon = out;
  float* out_loss  = out + (size_t)B_ * D_;
  float* out_dists = out + (size_t)B_ * D_ + 1;

  gumbel_vq_kernel<<<B_ / MTILE, 256, smem, stream>>>(
      targets, noise, cbb, cbt, out_recon, out_dists, acc);

  finalize_kernel<<<1, 64, 0, stream>>>(acc, out_loss);
}
